// LSTM_43327630082682
// MI455X (gfx1250) — matrix operand order repaired and checked by reading, not compiled, not run on hardware
//
#include <hip/hip_runtime.h>
#include <hip/hip_bf16.h>

typedef __attribute__((ext_vector_type(16))) __bf16 v16bf;
typedef __attribute__((ext_vector_type(8)))  float  v8f;

constexpr int kH = 512;   // hidden
constexpr int kT = 512;   // timesteps
constexpr int kB = 1024;  // batch
constexpr int kC = 10;    // classes

union V8F  { v8f v;  float f[8]; };
union V16B { v16bf v; unsigned int u[8]; };

// low 32 bits of the flat address of a __shared__ object == LDS byte offset
__device__ __forceinline__ unsigned int lds_off(const void* p) {
  return (unsigned int)(unsigned long long)p;
}
// async DMA: 16B per lane, global -> LDS, tracked by ASYNCcnt (no VGPR staging)
__device__ __forceinline__ void async_b128(unsigned int ldsoff, const void* g) {
  asm volatile("global_load_async_to_lds_b128 %0, %1, off"
               :: "v"(ldsoff), "v"(g) : "memory");
}

// ---------------- pack: W_*h (f32, [k][n]) -> WbT bf16 [g][n][k] ----------------
__global__ void lstm_pack_w(const float* __restrict__ Wg, const float* __restrict__ Wi,
                            const float* __restrict__ Wf, const float* __restrict__ Wo,
                            __bf16* __restrict__ WbT) {
  int idx = blockIdx.x * 256 + threadIdx.x;        // 4*512*512 threads total
  int g   = idx >> 18;
  int rem = idx & 0x3FFFF;
  int k   = rem >> 9;
  int n   = rem & 511;
  const float* W = (g == 0) ? Wg : (g == 1) ? Wi : (g == 2) ? Wf : Wo;
  WbT[((size_t)(g * kH + n)) * kH + k] = (__bf16)W[k * kH + n];
}

// ---------------- pack: x-coefficient vectors (W_*x is H x 1) and biases --------
__global__ void lstm_pack_vec(const float* __restrict__ Wgx, const float* __restrict__ Wix,
                              const float* __restrict__ Wfx, const float* __restrict__ Wox,
                              const float* __restrict__ bg,  const float* __restrict__ bi,
                              const float* __restrict__ bf,  const float* __restrict__ bo,
                              float* __restrict__ xw, float* __restrict__ bias) {
  int idx = blockIdx.x * 256 + threadIdx.x;        // 2048 threads
  int g = idx >> 9;
  int n = idx & 511;
  const float* Wx = (g == 0) ? Wgx : (g == 1) ? Wix : (g == 2) ? Wfx : Wox;
  const float* bv = (g == 0) ? bg  : (g == 1) ? bi  : (g == 2) ? bf  : bo;
  xw[idx]   = Wx[n];
  bias[idx] = bv[n];
}

// ---------------- init h0 (bf16) and c (f32) to zero ---------------------------
__global__ void lstm_init(__bf16* __restrict__ h0, float* __restrict__ c) {
  int idx = blockIdx.x * 256 + threadIdx.x;        // kB*kH threads
  h0[idx] = (__bf16)0.0f;
  c[idx]  = 0.0f;
}

// ---------------- one recurrent step -------------------------------------------
// Block: 256 threads = 8 waves. Block tile: 64 (batch) x 32 (hidden).
// Wave (wm in 0..3, wn in 0..1) owns a 16x16 output tile, 4 gate accumulators.
// LDS is double-buffered; staging uses global_load_async_to_lds_b128 so chunk
// i+1 streams in while chunk i's WMMAs execute.
__global__ __launch_bounds__(256) void lstm_step(
    const __bf16* __restrict__ hin, __bf16* __restrict__ hout,
    float* __restrict__ c, const float* __restrict__ x,
    const __bf16* __restrict__ WbT, const float* __restrict__ xw,
    const float* __restrict__ bias, int t) {
  // 20-dword row stride: 16 data dwords + 4 pad (bank-conflict-free, 16B aligned rows)
  __shared__ __align__(16) unsigned int ldsA[2][64 * 20];
  __shared__ __align__(16) unsigned int ldsB[2][4 * 32 * 20];

  const int tid  = threadIdx.x;
  const int lane = tid & 31;
  const int w    = tid >> 5;
  const int wm   = w & 3;
  const int wn   = w >> 2;
  const int half = lane >> 4;
  const int l15  = lane & 15;
  const int bRow = blockIdx.x * 64;
  const int nCol = blockIdx.y * 32;

  V8F acc[4];
#pragma unroll
  for (int g = 0; g < 4; ++g)
#pragma unroll
    for (int j = 0; j < 8; ++j) acc[g].f[j] = 0.0f;

  // per-thread staging assignment (16B chunks)
  const int arow = tid >> 2;                       // A: 0..63
  const int ach  = tid & 3;
  const int c0 = tid,        g0 = c0 >> 7, r0 = (c0 & 127) >> 2, ch0 = c0 & 3;
  const int c1 = tid + 256,  g1 = c1 >> 7, r1 = (c1 & 127) >> 2, ch1 = c1 & 3;
  const __bf16* aSrc  = hin + (size_t)(bRow + arow) * kH + ach * 8;
  const __bf16* bSrc0 = WbT + ((size_t)(g0 * kH + nCol + r0)) * kH + ch0 * 8;
  const __bf16* bSrc1 = WbT + ((size_t)(g1 * kH + nCol + r1)) * kH + ch1 * 8;
  unsigned int aDst[2], bDst0[2], bDst1[2];
#pragma unroll
  for (int buf = 0; buf < 2; ++buf) {
    aDst[buf]  = lds_off(&ldsA[buf][arow * 20 + ach * 4]);
    bDst0[buf] = lds_off(&ldsB[buf][(g0 * 32 + r0) * 20 + ch0 * 4]);
    bDst1[buf] = lds_off(&ldsB[buf][(g1 * 32 + r1) * 20 + ch1 * 4]);
  }

  // prologue: start chunk 0 into buffer 0
  async_b128(aDst[0], aSrc);
  async_b128(bDst0[0], bSrc0);
  async_b128(bDst1[0], bSrc1);

  for (int i = 0; i < kH / 32; ++i) {
    const int buf = i & 1;
    if (i < kH / 32 - 1) {
      const int k1 = (i + 1) * 32;                 // stream next chunk now
      async_b128(aDst[buf ^ 1],  aSrc  + k1);
      async_b128(bDst0[buf ^ 1], bSrc0 + k1);
      async_b128(bDst1[buf ^ 1], bSrc1 + k1);
      // 3 ops (next chunk) may stay in flight; current chunk's 3 are done
      asm volatile("s_wait_asynccnt 0x3" ::: "memory");
    } else {
      asm volatile("s_wait_asynccnt 0x0" ::: "memory");
    }
    __syncthreads();                               // chunk i visible to all waves

    // A fragment (contiguous K, matching B): lane<16 -> K 0..15
    V16B a;
    {
      int rb = (wm * 16 + l15) * 20 + half * 8;
#pragma unroll
      for (int j = 0; j < 4; ++j) {
        a.u[j]     = ldsA[buf][rb + j];
        a.u[4 + j] = ldsA[buf][rb + 4 + j];
      }
    }
    // B fragments per gate (lanes 0-15 hold K 0..15 of column n), 4 WMMAs
#pragma unroll
    for (int g = 0; g < 4; ++g) {
      V16B b;
      int rb = (g * 32 + wn * 16 + l15) * 20 + half * 8;
#pragma unroll
      for (int j = 0; j < 8; ++j) b.u[j] = ldsB[buf][rb + j];
      acc[g].v = __builtin_amdgcn_wmma_f32_16x16x32_bf16(
          false, a.v, false, b.v, (short)0, acc[g].v, false, false);
    }
    __syncthreads();                               // all reads of buf done before reuse
  }

  // epilogue: C/D layout -> lane row r maps to M = r + 8*half, N = l15
  const int n   = nCol + wn * 16 + l15;
  const float wx0 = xw[0 * kH + n], wx1 = xw[1 * kH + n], wx2 = xw[2 * kH + n], wx3 = xw[3 * kH + n];
  const float bb0 = bias[0 * kH + n], bb1 = bias[1 * kH + n], bb2 = bias[2 * kH + n], bb3 = bias[3 * kH + n];
#pragma unroll
  for (int r = 0; r < 8; ++r) {
    int b = bRow + wm * 16 + half * 8 + r;
    float xv = x[(size_t)b * kT + t];
    float pg = acc[0].f[r] + xv * wx0 + bb0;
    float pi = acc[1].f[r] + xv * wx1 + bb1;
    float pf = acc[2].f[r] + xv * wx2 + bb2;
    float po = acc[3].f[r] + xv * wx3 + bb3;
    float gg = tanhf(pg);
    float ii = 1.0f / (1.0f + expf(-pi));
    float ff = tanhf(pf);
    float oo = tanhf(po);
    size_t off = (size_t)b * kH + n;
    float cn = gg * ii + c[off] * ff;
    c[off]   = cn;
    hout[off] = (__bf16)(tanhf(cn) * oo);
  }
}

// ---------------- final projection: out = h_T @ W_ph.T + b_p --------------------
__global__ void lstm_proj(const __bf16* __restrict__ hT, const float* __restrict__ Wp,
                          const float* __restrict__ bp, float* __restrict__ out) {
  int b   = blockIdx.x;
  int cls = threadIdx.x;
  if (cls >= kC) return;
  float s = bp[cls];
  const __bf16* hr = hT + (size_t)b * kH;
  const float*  wr = Wp + (size_t)cls * kH;
  for (int n = 0; n < kH; ++n) s += (float)hr[n] * wr[n];
  out[b * kC + cls] = s;
}

extern "C" void kernel_launch(void* const* d_in, const int* in_sizes, int n_in,
                              void* d_out, int out_size, void* d_ws, size_t ws_size,
                              hipStream_t stream) {
  const float* x    = (const float*)d_in[0];
  const float* W_gx = (const float*)d_in[1];
  const float* W_gh = (const float*)d_in[2];
  const float* b_g  = (const float*)d_in[3];
  const float* W_ix = (const float*)d_in[4];
  const float* W_ih = (const float*)d_in[5];
  const float* b_i  = (const float*)d_in[6];
  const float* W_fx = (const float*)d_in[7];
  const float* W_fh = (const float*)d_in[8];
  const float* b_f  = (const float*)d_in[9];
  const float* W_ox = (const float*)d_in[10];
  const float* W_oh = (const float*)d_in[11];
  const float* b_o  = (const float*)d_in[12];
  const float* W_ph = (const float*)d_in[13];
  const float* b_p  = (const float*)d_in[14];
  float* out = (float*)d_out;

  char* ws = (char*)d_ws;
  __bf16* WbT  = (__bf16*)(ws);                  // 4*512*512 bf16 = 2 MB
  float*  xw   = (float*)(ws + 2097152);         // 2048 f32
  float*  bias = (float*)(ws + 2105344);         // 2048 f32
  __bf16* h0   = (__bf16*)(ws + 2113536);        // 1 MB
  __bf16* h1   = (__bf16*)(ws + 3162112);        // 1 MB
  float*  c    = (float*)(ws + 4210688);         // 2 MB   (total ~6 MB)

  lstm_pack_w<<<4096, 256, 0, stream>>>(W_gh, W_ih, W_fh, W_oh, WbT);
  lstm_pack_vec<<<8, 256, 0, stream>>>(W_gx, W_ix, W_fx, W_ox, b_g, b_i, b_f, b_o, xw, bias);
  lstm_init<<<2048, 256, 0, stream>>>(h0, c);

  dim3 grid(kB / 64, kH / 32);  // 16 x 16 blocks
  for (int t = 0; t < kT; ++t) {
    const __bf16* hi = (t & 1) ? h1 : h0;
    __bf16*       ho = (t & 1) ? h0 : h1;
    lstm_step<<<grid, 256, 0, stream>>>(hi, ho, c, x, WbT, xw, bias, t);
  }
  // t = 511 (odd) wrote h0 -> final hidden state is h0
  lstm_proj<<<kB, 32, 0, stream>>>(h0, W_ph, b_p, out);
}
